// ViewPrompter_30863634989683
// MI455X (gfx1250) — compile-verified, matrix operand order checked by reading, not run on hardware
//
#include <hip/hip_runtime.h>

// ---------------- problem dims ----------------
#define EPSF    1e-5f
#define SLOPEF  0.01f
#define BATCH   64
#define CIN     3
#define HIN     224
#define C1      96      // stem out channels
#define H1      56      // stem out spatial
#define HO      14      // dcn out spatial
#define KKN     16      // 4x4 kernel
#define COUT    768
#define KDIM    1536    // 96*16
#define M_TOT   12544   // 64*196
#define NKT     (KDIM / 32)

typedef __attribute__((ext_vector_type(16))) _Float16 v16h;
typedef __attribute__((ext_vector_type(8)))  _Float16 v8h;
typedef __attribute__((ext_vector_type(8)))  float    v8f;

// ---------------- kernel 1: stem conv + BN1 + leaky ----------------
// h[b][oc][y][x] ; 4x4 stride-4 VALID conv over (3,224,224)
__global__ __launch_bounds__(256) void k_stem(
    const float* __restrict__ x, const float* __restrict__ w,
    const float* __restrict__ bias,
    const float* __restrict__ g, const float* __restrict__ bb,
    const float* __restrict__ mmn, const float* __restrict__ vv,
    float* __restrict__ h) {
  int idx = blockIdx.x * 256 + threadIdx.x;       // total = 64*96*56*56
  int xw  = idx % H1;
  int yh  = (idx / H1) % H1;
  int oc  = (idx / (H1 * H1)) % C1;
  int b   = idx / (H1 * H1 * C1);
  float acc = bias[oc];
  const float* wp = w + oc * 48;
#pragma unroll
  for (int c = 0; c < CIN; ++c) {
    const float* xp = x + ((size_t)(b * CIN + c) * HIN + yh * 4) * HIN + xw * 4;
#pragma unroll
    for (int r = 0; r < 4; ++r) {
      float4 v = *(const float4*)(xp + r * HIN);
      const float* ww = wp + c * 16 + r * 4;
      acc += v.x * ww[0] + v.y * ww[1] + v.z * ww[2] + v.w * ww[3];
    }
  }
  float inv = g[oc] * rsqrtf(vv[oc] + EPSF);
  float val = acc * inv + (bb[oc] - mmn[oc] * inv);
  h[idx] = val >= 0.f ? val : SLOPEF * val;
}

// ---------------- kernel 2: convert dcn weights to f16 ----------------
__global__ __launch_bounds__(256) void k_cvt(const float* __restrict__ w,
                                             _Float16* __restrict__ o, int n) {
  int i = blockIdx.x * 256 + threadIdx.x;
  if (i < n) o[i] = (_Float16)w[i];
}

// ---------------- kernel 3: offset conv (96->32, 4x4/s4) ----------------
// offs layout: [m][oc], m = b*196 + ho*14 + wo, oc = kk*2 + {y,x}
__global__ __launch_bounds__(256) void k_offconv(
    const float* __restrict__ h, const float* __restrict__ ow,
    const float* __restrict__ ob, float* __restrict__ offs) {
  __shared__ float patch[8 * KDIM];               // 48 KB
  int m0 = blockIdx.x * 8;
  int tid = threadIdx.x;
  for (int j = tid; j < 8 * KDIM; j += 256) {
    int p = j / KDIM, k = j - p * KDIM;
    int c = k >> 4, kh = (k >> 2) & 3, kw = k & 3;
    int m = m0 + p;
    int b = m / 196, hw = m - b * 196;
    int ho = hw / 14, wo = hw - ho * 14;
    patch[j] = h[((size_t)(b * C1 + c) * H1 + ho * 4 + kh) * H1 + wo * 4 + kw];
  }
  __syncthreads();
  int p = tid >> 5, oc = tid & 31;
  const float* pp = patch + p * KDIM;
  const float* wp = ow + oc * KDIM;
  float acc = ob[oc];
#pragma unroll 8
  for (int k = 0; k < KDIM; ++k) acc += pp[k] * wp[k];
  offs[(size_t)(m0 + p) * 32 + oc] = acc;
}

// ---------------- kernel 4: bilinear sampling -> f16 im2col A ----------------
__device__ __forceinline__ float fetch2d(const float* p, int y, int x) {
  return (y >= 0 && y < H1 && x >= 0 && x < H1) ? p[y * H1 + x] : 0.f;
}

__global__ __launch_bounds__(256) void k_sample(
    const float* __restrict__ h, const float* __restrict__ offs,
    _Float16* __restrict__ A) {
  int idx = blockIdx.x * 256 + threadIdx.x;       // total = M_TOT*16
  int kk = idx & 15, m = idx >> 4;
  int b = m / 196, hw = m - b * 196;
  int ho = hw / 14, wo = hw - ho * 14;
  int kh = kk >> 2, kw = kk & 3;
  float py = (float)(ho * 4 + kh) + offs[(size_t)m * 32 + kk * 2 + 0];
  float px = (float)(wo * 4 + kw) + offs[(size_t)m * 32 + kk * 2 + 1];
  float y0f = floorf(py), x0f = floorf(px);
  float wy = py - y0f, wx = px - x0f;
  int y0 = (int)y0f, x0 = (int)x0f;
  float w00 = (1.f - wy) * (1.f - wx), w01 = (1.f - wy) * wx;
  float w10 = wy * (1.f - wx),         w11 = wy * wx;
  const float* hb = h + (size_t)b * C1 * H1 * H1;
  _Float16* ap = A + (size_t)m * KDIM + kk;
  for (int c = 0; c < C1; ++c) {
    const float* pl = hb + (size_t)c * H1 * H1;
    float s = fetch2d(pl, y0, x0)         * w00
            + fetch2d(pl, y0, x0 + 1)     * w01
            + fetch2d(pl, y0 + 1, x0)     * w10
            + fetch2d(pl, y0 + 1, x0 + 1) * w11;
    ap[c * 16] = (_Float16)s;
  }
}

// ---------------- kernel 5: WMMA GEMM + bias + BN2 + leaky -> d_out ----------------
#define LDSK 40   // padded K-stride (halves); row stride 80B keeps 16B alignment

// Async direct global->LDS copy (ASYNCcnt path, GLOBAL_LOAD_ASYNC_TO_LDS_B128).
// LDS operand is the wave-relative LDS byte offset = low 32 bits of the generic
// pointer (flat aperture rule).
__device__ __forceinline__ void async_ldg_b128(_Float16* lds, const _Float16* g) {
  unsigned int l = (unsigned int)(unsigned long long)lds;
  asm volatile("global_load_async_to_lds_b128 %0, %1, off"
               :: "v"(l), "v"(g) : "memory");
}

__device__ __forceinline__ void stage_async(_Float16* asb, _Float16* bsb,
                                            const _Float16* gA, const _Float16* gB,
                                            int rowL, int colh, int kt) {
  _Float16* la = asb + rowL * LDSK + colh;
  _Float16* lb = bsb + rowL * LDSK + colh;
  const _Float16* pa = gA + kt * 32;
  const _Float16* pb = gB + kt * 32;
  async_ldg_b128(la,     pa);
  async_ldg_b128(la + 8, pa + 8);
  async_ldg_b128(lb,     pb);
  async_ldg_b128(lb + 8, pb + 8);
}

__device__ __forceinline__ v16h frag_from_lds(const _Float16* base, int row, int kh) {
  const _Float16* p = base + row * LDSK + kh;
  v8h lo = *(const v8h*)(p);
  v8h hi = *(const v8h*)(p + 16);
  return __builtin_shufflevector(lo, hi, 0, 1, 2, 3, 4, 5, 6, 7,
                                         8, 9, 10, 11, 12, 13, 14, 15);
}

__global__ __launch_bounds__(256) void k_gemm(
    const _Float16* __restrict__ A, const _Float16* __restrict__ Bw,
    const float* __restrict__ dbias,
    const float* __restrict__ g, const float* __restrict__ bb,
    const float* __restrict__ mmn, const float* __restrict__ vv,
    float* __restrict__ out) {
  __shared__ _Float16 As[2][128 * LDSK];          // 2 x 10 KB
  __shared__ _Float16 Bs[2][128 * LDSK];          // 2 x 10 KB
  int tid = threadIdx.x;
  int bn = blockIdx.x;                 // 0..5   (N tiles of 128)
  int bm = blockIdx.y;                 // 0..97  (M tiles of 128)
  int lane = tid & 31, wave = tid >> 5;
  int wm = wave & 1, wn = wave >> 1;   // 2x4 wave grid -> 64x32 per wave
  int rowL = tid >> 1;                 // staging row 0..127
  int colh = (tid & 1) * 16;           // staging col (halves)
  const _Float16* gA = A  + (size_t)(bm * 128 + rowL) * KDIM + colh;
  const _Float16* gB = Bw + (size_t)(bn * 128 + rowL) * KDIM + colh;
  v8f acc[4][2] = {};
  int khalf = (lane < 16) ? 0 : 8;     // ISA 16-bit A/B lane->K mapping

  stage_async(As[0], Bs[0], gA, gB, rowL, colh, 0);   // prologue: stage kt=0

  for (int kt = 0; kt < NKT; ++kt) {
    int cur = kt & 1;
    if (kt + 1 < NKT) {
      // issue next stage into the other buffer, then wait only for the
      // current stage (4 newer async ops may remain outstanding)
      stage_async(As[cur ^ 1], Bs[cur ^ 1], gA, gB, rowL, colh, kt + 1);
      asm volatile("s_wait_asynccnt 0x4" ::: "memory");
    } else {
      asm volatile("s_wait_asynccnt 0x0" ::: "memory");
    }
    __syncthreads();

    v16h af[4], bf[2];
#pragma unroll
    for (int i = 0; i < 4; ++i)
      af[i] = frag_from_lds(As[cur], wm * 64 + i * 16 + (lane & 15), khalf);
#pragma unroll
    for (int j = 0; j < 2; ++j)
      bf[j] = frag_from_lds(Bs[cur], wn * 32 + j * 16 + (lane & 15), khalf);
#pragma unroll
    for (int i = 0; i < 4; ++i)
#pragma unroll
      for (int j = 0; j < 2; ++j)
        acc[i][j] = __builtin_amdgcn_wmma_f32_16x16x32_f16(
            false, af[i], false, bf[j], (short)0, acc[i][j], false, false);

    __syncthreads();   // compute done before this buffer is re-staged (kt+2)
  }

  // epilogue: bias + BN2 + leaky, direct store to (B, HW, C) = row-major MxN
#pragma unroll
  for (int j = 0; j < 2; ++j) {
    int gn = bn * 128 + wn * 32 + j * 16 + (lane & 15);
    float inv = g[gn] * rsqrtf(vv[gn] + EPSF);
    float shf = bb[gn] - mmn[gn] * inv;
    float bv = dbias[gn];
#pragma unroll
    for (int i = 0; i < 4; ++i) {
      int gm0 = bm * 128 + wm * 64 + i * 16 + ((lane >> 4) << 3);
#pragma unroll
      for (int e = 0; e < 8; ++e) {
        float val = (acc[i][j][e] + bv) * inv + shf;
        val = val >= 0.f ? val : SLOPEF * val;
        out[(size_t)(gm0 + e) * COUT + gn] = val;
      }
    }
  }
}

// ---------------- launch ----------------
extern "C" void kernel_launch(void* const* d_in, const int* in_sizes, int n_in,
                              void* d_out, int out_size, void* d_ws, size_t ws_size,
                              hipStream_t stream) {
  (void)in_sizes; (void)n_in; (void)out_size; (void)ws_size;
  const float* x      = (const float*)d_in[0];
  const float* stem_w = (const float*)d_in[1];
  const float* stem_b = (const float*)d_in[2];
  const float* bn1_g  = (const float*)d_in[3];
  const float* bn1_b  = (const float*)d_in[4];
  const float* bn1_m  = (const float*)d_in[5];
  const float* bn1_v  = (const float*)d_in[6];
  const float* off_w  = (const float*)d_in[7];
  const float* off_b  = (const float*)d_in[8];
  const float* dcn_w  = (const float*)d_in[9];
  const float* dcn_b  = (const float*)d_in[10];
  const float* bn2_g  = (const float*)d_in[11];
  const float* bn2_b  = (const float*)d_in[12];
  const float* bn2_m  = (const float*)d_in[13];
  const float* bn2_v  = (const float*)d_in[14];

  // workspace carve-up (all 16B aligned)
  char* ws = (char*)d_ws;
  const size_t H_BYTES = (size_t)BATCH * C1 * H1 * H1 * 4;        // 77,070,336
  const size_t A_BYTES = (size_t)M_TOT * KDIM * 2;                // 38,535,168
  const size_t W_BYTES = (size_t)COUT * KDIM * 2;                 //  2,359,296
  float*    h    = (float*)ws;
  _Float16* Amat = (_Float16*)(ws + H_BYTES);
  _Float16* Wf16 = (_Float16*)(ws + H_BYTES + A_BYTES);
  float*    offs = (float*)(ws + H_BYTES + A_BYTES + W_BYTES);    // 1,605,632

  // 1) stem conv + BN1 + leaky
  k_stem<<<(BATCH * C1 * H1 * H1) / 256, 256, 0, stream>>>(
      x, stem_w, stem_b, bn1_g, bn1_b, bn1_m, bn1_v, h);
  // 2) dcn weights -> f16
  k_cvt<<<(COUT * KDIM + 255) / 256, 256, 0, stream>>>(dcn_w, Wf16, COUT * KDIM);
  // 3) offset conv
  k_offconv<<<M_TOT / 8, 256, 0, stream>>>(h, off_w, off_b, offs);
  // 4) bilinear sampling -> f16 im2col A
  k_sample<<<(M_TOT * KKN) / 256, 256, 0, stream>>>(h, offs, Amat);
  // 5) WMMA GEMM (12544x768x1536) + bias + BN2 + leaky -> d_out
  k_gemm<<<dim3(COUT / 128, M_TOT / 128), 256, 0, stream>>>(
      Amat, Wf16, dcn_b, bn2_g, bn2_b, bn2_m, bn2_v, (float*)d_out);
}
